// FuncRepresentation_79697413144920
// MI455X (gfx1250) — matrix operand order repaired and checked by reading, not compiled, not run on hardware
//
#include <hip/hip_runtime.h>

// CDNA5 (gfx1250, wave32) WMMA implementation of the RBF functional
// representation kernel.
//
//   dens[b,m,p] = sum_t mask[b,m,t] * exp(-(g_p - x[b,t])^2 * inv0)
//   conv[b,m,p] = sum_t    y[b,m,t] * exp(-(g_p - x[b,t])^2 * inv1)
//   out[b,p,2m+0] = dens ; out[b,p,2m+1] = conv / (dens + 1e-8)
//
// One wave per (b, 16-wide p tile); K-loop over T=512 in 16 chunks of 32
// with v_wmma_f32_16x16x32_f16. The exp kernel (B operand) is built per-lane
// with v_exp_f32 via the raw exp2 builtin (log2e pre-folded into the scale,
// saving one v_mul per exp). Loads are software-pipelined through two
// register chunk buffers so global latency hides under the exp/VALU work.

typedef __attribute__((ext_vector_type(16))) _Float16 v16h;
typedef __attribute__((ext_vector_type(8)))  float    v8f;

#define NB 32
#define NM 16
#define NT 512
#define NP 600
#define PTILES 38              // ceil(600 / 16)
#define NWAVES (NB * PTILES)   // 1216

struct ChunkBuf {
    float4 yv[4];   // A operand source: y row, 16 floats
    int4   mv[4];   // A operand source: mask row, 16 ints
    float4 xv[4];   // B operand source: x times, 16 floats
};

__device__ __forceinline__ void load_chunk(ChunkBuf& c,
        const float* __restrict__ yrow, const int* __restrict__ mrow,
        const float* __restrict__ xrow, int t0, int hi)
{
    const int b1 = t0 + hi * 8;        // A f16 16x32 layout: lane half picks
    const int b2 = t0 + 16 + hi * 8;   // K = [b1..b1+7] and [b2..b2+7]
    const int tb = t0 + hi * 16;       // B f16 32x16 layout: same K order as A (b1, b2)
    c.yv[0] = *(const float4*)(yrow + b1);
    c.yv[1] = *(const float4*)(yrow + b1 + 4);
    c.yv[2] = *(const float4*)(yrow + b2);
    c.yv[3] = *(const float4*)(yrow + b2 + 4);
    c.mv[0] = *(const int4*)(mrow + b1);
    c.mv[1] = *(const int4*)(mrow + b1 + 4);
    c.mv[2] = *(const int4*)(mrow + b2);
    c.mv[3] = *(const int4*)(mrow + b2 + 4);
    c.xv[0] = *(const float4*)(xrow + b1);
    c.xv[1] = *(const float4*)(xrow + b1 + 4);
    c.xv[2] = *(const float4*)(xrow + b2);
    c.xv[3] = *(const float4*)(xrow + b2 + 4);
}

__device__ __forceinline__ void process_chunk(const ChunkBuf& c,
        float gridp, float c0e, float c1e, v8f& acc0, v8f& acc1)
{
    float ytmp[16]; int mtmp[16]; float xtmp[16];
    *(float4*)&ytmp[0]  = c.yv[0]; *(float4*)&ytmp[4]  = c.yv[1];
    *(float4*)&ytmp[8]  = c.yv[2]; *(float4*)&ytmp[12] = c.yv[3];
    *(int4*)&mtmp[0]    = c.mv[0]; *(int4*)&mtmp[4]    = c.mv[1];
    *(int4*)&mtmp[8]    = c.mv[2]; *(int4*)&mtmp[12]   = c.mv[3];
    *(float4*)&xtmp[0]  = c.xv[0]; *(float4*)&xtmp[4]  = c.xv[1];
    *(float4*)&xtmp[8]  = c.xv[2]; *(float4*)&xtmp[12] = c.xv[3];

    v16h a_y, a_d, b0, b1;
    #pragma unroll
    for (int e = 0; e < 16; ++e) {
        a_y[e] = (_Float16)ytmp[e];
        a_d[e] = (_Float16)(float)mtmp[e];
        const float d  = gridp - xtmp[e];
        const float d2 = d * d;
        // exp(-d2*inv) == exp2(d2 * (-inv*log2e)); raw v_exp_f32, no extra mul
        b0[e] = (_Float16)__builtin_amdgcn_exp2f(d2 * c0e);
        b1[e] = (_Float16)__builtin_amdgcn_exp2f(d2 * c1e);
    }
    acc0 = __builtin_amdgcn_wmma_f32_16x16x32_f16(
               false, a_d, false, b0, (short)0, acc0, false, false);
    acc1 = __builtin_amdgcn_wmma_f32_16x16x32_f16(
               false, a_y, false, b1, (short)0, acc1, false, false);
}

__global__ __launch_bounds__(256)
void func_repr_wmma_kernel(const float* __restrict__ y,     // [B, M, T]
                           const float* __restrict__ x,     // [B, T]
                           const int*   __restrict__ mask,  // [B, M, T]
                           const float* __restrict__ sigma, // [2]
                           float*       __restrict__ out)   // [B, P, 2*M]
{
    const int wave = (int)((blockIdx.x * blockDim.x + threadIdx.x) >> 5);
    if (wave >= NWAVES) return;                 // wave-uniform: EXEC stays all-1
    const int lane = (int)(threadIdx.x & 31);

    const int b  = wave / PTILES;
    const int pt = wave % PTILES;
    const int p0 = pt * 16;

    const int col = lane & 15;                  // B column N / A row m
    const int hi  = lane >> 4;                  // lane half selects K sub-range

    // inv_half = 0.5 / sigma^2, with log2(e) folded in for the exp2 path
    const float s0 = sigma[0], s1 = sigma[1];
    const float LOG2E = 1.4426950408889634f;
    const float c0e = -(0.5f / (s0 * s0)) * LOG2E;
    const float c1e = -(0.5f / (s1 * s1)) * LOG2E;

    // grid = linspace(-1, 49, 600): step = 50/599
    const int   p     = p0 + col;
    const float gridp = -1.0f + (float)p * (50.0f / 599.0f);

    const float* yrow = y    + (size_t)(b * NM + col) * NT; // A row m = col
    const int*   mrow = mask + (size_t)(b * NM + col) * NT;
    const float* xrow = x    + (size_t)b * NT;

    v8f acc0 = {};   // density accumulator (f32)
    v8f acc1 = {};   // conv accumulator (f32)

    // Software pipeline: two chunk buffers, loads one chunk ahead of use.
    ChunkBuf bufA, bufB;
    load_chunk(bufA, yrow, mrow, xrow, 0,  hi);
    load_chunk(bufB, yrow, mrow, xrow, 32, hi);

    #pragma unroll 1
    for (int kc = 0; kc < 14; kc += 2) {        // processes chunks 0..13
        process_chunk(bufA, gridp, c0e, c1e, acc0, acc1);
        load_chunk(bufA, yrow, mrow, xrow, (kc + 2) * 32, hi);
        process_chunk(bufB, gridp, c0e, c1e, acc0, acc1);
        load_chunk(bufB, yrow, mrow, xrow, (kc + 3) * 32, hi);
    }
    process_chunk(bufA, gridp, c0e, c1e, acc0, acc1);   // chunk 14
    process_chunk(bufB, gridp, c0e, c1e, acc0, acc1);   // chunk 15

    // C/D 16x16 f32 layout: VGPR j -> row M = j + 8*hi, col N = col.
    if (p < NP) {
        float* orow = out + ((size_t)b * NP + p) * (2 * NM);
        #pragma unroll
        for (int j = 0; j < 8; ++j) {
            const int   m    = j + hi * 8;
            const float dens = acc0[j];
            const float conv = acc1[j];
            float2 v;
            v.x = dens;
            v.y = conv / (dens + 1e-8f);
            *(float2*)(orow + 2 * m) = v;       // 8-byte aligned pair store
        }
    }
}

extern "C" void kernel_launch(void* const* d_in, const int* in_sizes, int n_in,
                              void* d_out, int out_size, void* d_ws, size_t ws_size,
                              hipStream_t stream) {
    (void)in_sizes; (void)n_in; (void)d_ws; (void)ws_size; (void)out_size;
    const float* y     = (const float*)d_in[0];   // [32,16,512] f32
    const float* x     = (const float*)d_in[1];   // [32,512]    f32
    const int*   mask  = (const int*)  d_in[2];   // [32,16,512] i32
    const float* sigma = (const float*)d_in[3];   // [2]         f32
    float*       out   = (float*)d_out;           // [32,600,32] f32

    const int threads = 256;                       // 8 waves per block
    const int blocks  = (NWAVES * 32 + threads - 1) / threads;  // 152
    func_repr_wmma_kernel<<<blocks, threads, 0, stream>>>(y, x, mask, sigma, out);
}
